// GraphEncoder_31310311587887
// MI455X (gfx1250) — matrix operand order repaired and checked by reading, not compiled, not run on hardware
//
#include <hip/hip_runtime.h>

// ---------------- problem constants ----------------
#define NB      8192        // batch
#define NT      4           // edge types
#define NNEI    10          // neighbors
#define ND      128         // embed dim
#define NA      32          // attention dim
#define NTILES  516         // >= 512 + 3 padded tiles (+1 slack)
#define ORDLEN  (NTILES*16) // 8256

// ---------------- workspace layout (bytes) ----------------
#define S1P_OFF 0           // bf16 s1 packed: 4*128*32*2   = 32768
#define WP_OFF  32768       // bf16 W  packed: 4*128*128*2  = 131072
#define CUR_OFF 163840      // 4 x uint cursors
#define TT_OFF  163872      // int tile_type[516]
#define ORD_OFF 165952      // int order[8256]
// total ~198976 bytes

typedef float v4f  __attribute__((ext_vector_type(4)));
typedef float v8f  __attribute__((ext_vector_type(8)));
typedef __attribute__((ext_vector_type(16))) __bf16 v16bf;

__device__ __forceinline__ unsigned short f2bf(float f) {
    unsigned u = __builtin_bit_cast(unsigned, f);
    u += 0x7FFFu + ((u >> 16) & 1u);           // round-to-nearest-even
    return (unsigned short)(u >> 16);
}
__device__ __forceinline__ float bf2f(unsigned short h) {
    return __builtin_bit_cast(float, ((unsigned)h) << 16);
}

// ---------------------------------------------------------------------------
// Kernel 1: repack f32 weights into bf16 WMMA B-matrix lane layout.
// B order used here (contiguous k, same as A loads; K=32 per kb): lanes 0-15 hold K=0..15 of column n=lane,
// lanes 16-31 hold K=16..31 of column n=lane-16; chunk c (16B) = 8 K values.
// Dest ushort index == flat thread id by construction.
// ---------------------------------------------------------------------------
__global__ void pack_weights(const float* __restrict__ s1,
                             const float* __restrict__ w,
                             unsigned short* __restrict__ s1p,
                             unsigned short* __restrict__ wp) {
    int tid = blockIdx.x * blockDim.x + threadIdx.x;
    if (tid < 16384) {   // s1: t:2 kb:2 nb:1 lane:5 c:1 j:3
        int j = tid & 7, c = (tid >> 3) & 1, lane = (tid >> 4) & 31;
        int nb = (tid >> 9) & 1, kb = (tid >> 10) & 3, t = (tid >> 12) & 3;
        int K = kb * 32 + ((lane < 16) ? 0 : 16) + c * 8 + j;
        int n = nb * 16 + (lane & 15);
        s1p[tid] = f2bf(s1[(t * ND + K) * NA + n]);
    }
    int id2 = tid - 16384;
    if (id2 >= 0 && id2 < 65536) { // W: t:2 kb:2 nb:3 lane:5 c:1 j:3
        int j = id2 & 7, c = (id2 >> 3) & 1, lane = (id2 >> 4) & 31;
        int nb = (id2 >> 9) & 7, kb = (id2 >> 12) & 3, t = (id2 >> 14) & 3;
        int K = kb * 32 + ((lane < 16) ? 0 : 16) + c * 8 + j;
        int e = nb * 16 + (lane & 15);
        wp[id2] = f2bf(w[(t * ND + K) * ND + e]);
    }
}

// ---------------------------------------------------------------------------
// Kernel 2: histogram of node_types, padded segment offsets, tile->type map.
// ---------------------------------------------------------------------------
__global__ void hist_kernel(const int* __restrict__ node_types,
                            unsigned* __restrict__ cursors,
                            int* __restrict__ tile_type,
                            int* __restrict__ order) {
    __shared__ int cnt[NT];
    int tid = threadIdx.x;
    if (tid < NT) cnt[tid] = 0;
    for (int i = tid; i < ORDLEN; i += 256) order[i] = -1;
    for (int i = tid; i < NTILES; i += 256) tile_type[i] = 0;
    __syncthreads();
    for (int i = tid; i < NB; i += 256) atomicAdd(&cnt[node_types[i]], 1);
    __syncthreads();
    if (tid == 0) {
        int off = 0, ti = 0;
        for (int t = 0; t < NT; ++t) {
            cursors[t] = (unsigned)off;
            int ntile = (cnt[t] + 15) >> 4;
            for (int k = 0; k < ntile; ++k) tile_type[ti++] = t;
            off += ntile * 16;
        }
    }
}

// ---------------------------------------------------------------------------
// Kernel 3: scatter sample ids into their type segment (order within a
// segment is irrelevant: every per-sample result is independent).
// ---------------------------------------------------------------------------
__global__ void scatter_kernel(const int* __restrict__ node_types,
                               unsigned* __restrict__ cursors,
                               int* __restrict__ order) {
    int b = blockIdx.x * blockDim.x + threadIdx.x;
    if (b < NB) {
        unsigned pos = atomicAdd(&cursors[node_types[b]], 1u);
        order[pos] = b;
    }
}

// ---------------------------------------------------------------------------
// Kernel 4: one 16-sample, type-uniform tile per block. 4 waves.
// ---------------------------------------------------------------------------
__launch_bounds__(128)
__global__ void gatne_tile_kernel(const int*   __restrict__ inputs,
                                  const int*   __restrict__ node_neigh,
                                  const float* __restrict__ node_emb,
                                  const float* __restrict__ nte,
                                  const float* __restrict__ s2,
                                  const unsigned short* __restrict__ s1p,
                                  const unsigned short* __restrict__ wp,
                                  const int*   __restrict__ tile_type,
                                  const int*   __restrict__ order,
                                  float*       __restrict__ out) {
    // bf16 A tiles in LDS, row stride 136 elems (272B) -> bank-conflict-free
    __shared__ unsigned short Ebf[64 * 136];   // rows r = t*16 + b, K = 128
    __shared__ unsigned short Agg[16 * 136];   // rows b, K = 128
    __shared__ float scores[NT][16];
    __shared__ float att[NT][16];
    __shared__ float normsq[16];
    __shared__ int   smp[16];

    const int tid  = threadIdx.x;
    const int wave = tid >> 5;
    const int lane = tid & 31;
    const int tile = blockIdx.x;
    const int type = tile_type[tile];

    if (tid < 16) smp[tid] = order[tile * 16 + tid];
    __syncthreads();

    // ---- Phase 1: neighbor gather + sum (wave w handles edge-type t=w) ----
    {
        const int t = wave;
        for (int b = 0; b < 16; ++b) {
            int s = smp[b];
            v4f acc = {0.f, 0.f, 0.f, 0.f};
            if (s >= 0) {
                const int* nidx = node_neigh + (s * NT + t) * NNEI;
                #pragma unroll
                for (int n = 0; n < NNEI; ++n) {
                    long long node = nidx[n];
                    const v4f* row = (const v4f*)(nte + (node * NT + t) * ND);
                    acc += row[lane];                    // global_load_b128
                }
            }
            unsigned lo = (unsigned)f2bf(acc.x) | ((unsigned)f2bf(acc.y) << 16);
            unsigned hi = (unsigned)f2bf(acc.z) | ((unsigned)f2bf(acc.w) << 16);
            uint2 pk; pk.x = lo; pk.y = hi;
            *(uint2*)&Ebf[(t * 16 + b) * 136 + lane * 4] = pk;
        }
    }
    __syncthreads();

    // ---- Phase 2: scores = reduce_a( tanh(E @ S1) * s2 ), via bf16 WMMA ----
    {
        union ABu { uint4 q[2]; v16bf v; };
        v8f acc0 = {0,0,0,0,0,0,0,0};
        v8f acc1 = {0,0,0,0,0,0,0,0};
        const int half8 = (lane < 16) ? 0 : 16;           // A layout K offset
        const int rrow  = wave * 16 + (lane & 15);
        const uint4* bq = (const uint4*)s1p;
        #pragma unroll
        for (int kb = 0; kb < 4; ++kb) {
            ABu a;
            const unsigned short* ap = &Ebf[rrow * 136 + kb * 32 + half8];
            a.q[0] = *(const uint4*)(ap);                 // ds_load_b128
            a.q[1] = *(const uint4*)(ap + 8);
            ABu b0, b1;
            int bi0 = (((type * 4 + kb) * 2 + 0) * 32 + lane) * 2;
            b0.q[0] = bq[bi0];      b0.q[1] = bq[bi0 + 1];
            b1.q[0] = bq[bi0 + 64]; b1.q[1] = bq[bi0 + 65];
            acc0 = __builtin_amdgcn_wmma_f32_16x16x32_bf16(false, a.v, false, b0.v,
                                                           (short)0, acc0, false, false);
            acc1 = __builtin_amdgcn_wmma_f32_16x16x32_bf16(false, a.v, false, b1.v,
                                                           (short)0, acc1, false, false);
        }
        const float s2a = s2[type * NA + (lane & 15)];
        const float s2b = s2[type * NA + 16 + (lane & 15)];
        #pragma unroll
        for (int i = 0; i < 8; ++i) {
            float p = tanhf(acc0[i]) * s2a + tanhf(acc1[i]) * s2b;
            p += __shfl_xor(p, 1, 32);
            p += __shfl_xor(p, 2, 32);
            p += __shfl_xor(p, 4, 32);
            p += __shfl_xor(p, 8, 32);
            if ((lane & 15) == 0) scores[wave][i + ((lane >> 4) << 3)] = p;
        }
    }
    __syncthreads();

    // ---- Phase 3: softmax over 4 types + f32 aggregation, bf16 store ----
    if (tid < 16) {
        float a0 = scores[0][tid], a1 = scores[1][tid];
        float a2 = scores[2][tid], a3 = scores[3][tid];
        float m = fmaxf(fmaxf(a0, a1), fmaxf(a2, a3));
        float e0 = expf(a0 - m), e1 = expf(a1 - m);
        float e2 = expf(a2 - m), e3 = expf(a3 - m);
        float inv = 1.0f / (e0 + e1 + e2 + e3);
        att[0][tid] = e0 * inv; att[1][tid] = e1 * inv;
        att[2][tid] = e2 * inv; att[3][tid] = e3 * inv;
        normsq[tid] = 0.f;
    }
    __syncthreads();
    {
        const int d = tid;                                 // 0..127
        for (int b = 0; b < 16; ++b) {
            float a = 0.f;
            #pragma unroll
            for (int t = 0; t < NT; ++t)
                a += att[t][b] * bf2f(Ebf[(t * 16 + b) * 136 + d]);
            Agg[b * 136 + d] = f2bf(a);
        }
    }
    __syncthreads();

    // ---- Phase 4: out = normalize(base + agg @ W), via bf16 WMMA ----
    {
        union ABu { uint4 q[2]; v16bf v; };
        v8f acc0 = {0,0,0,0,0,0,0,0};
        v8f acc1 = {0,0,0,0,0,0,0,0};
        const int half8 = (lane < 16) ? 0 : 16;
        const int arow  = lane & 15;
        const int nb0 = wave * 2, nb1 = wave * 2 + 1;      // wave owns 2 N-tiles
        const uint4* bq = (const uint4*)wp;
        #pragma unroll
        for (int kb = 0; kb < 4; ++kb) {
            ABu a;
            const unsigned short* ap = &Agg[arow * 136 + kb * 32 + half8];
            a.q[0] = *(const uint4*)(ap);
            a.q[1] = *(const uint4*)(ap + 8);
            ABu b0, b1;
            int bi = (((type * 4 + kb) * 8 + nb0) * 32 + lane) * 2;
            b0.q[0] = bq[bi];      b0.q[1] = bq[bi + 1];
            b1.q[0] = bq[bi + 64]; b1.q[1] = bq[bi + 65];
            acc0 = __builtin_amdgcn_wmma_f32_16x16x32_bf16(false, a.v, false, b0.v,
                                                           (short)0, acc0, false, false);
            acc1 = __builtin_amdgcn_wmma_f32_16x16x32_bf16(false, a.v, false, b1.v,
                                                           (short)0, acc1, false, false);
        }
        const int col = lane & 15;
        const int mh  = (lane >> 4) << 3;
        #pragma unroll
        for (int i = 0; i < 8; ++i) {                      // + base embedding
            int m = i + mh, s = smp[m];
            if (s >= 0) {
                long long node = inputs[s];
                acc0[i] += node_emb[node * ND + nb0 * 16 + col];
                acc1[i] += node_emb[node * ND + nb1 * 16 + col];
            }
        }
        #pragma unroll
        for (int i = 0; i < 8; ++i) {                      // row ||.||^2
            float ss = acc0[i] * acc0[i] + acc1[i] * acc1[i];
            ss += __shfl_xor(ss, 1, 32);
            ss += __shfl_xor(ss, 2, 32);
            ss += __shfl_xor(ss, 4, 32);
            ss += __shfl_xor(ss, 8, 32);
            if ((lane & 15) == 0) atomicAdd(&normsq[i + mh], ss);  // ds_add_f32
        }
        __syncthreads();
        #pragma unroll
        for (int i = 0; i < 8; ++i) {
            int m = i + mh, s = smp[m];
            if (s >= 0) {
                float scale = 1.0f / fmaxf(sqrtf(normsq[m]), 1e-12f);
                out[(long long)s * ND + nb0 * 16 + col] = acc0[i] * scale;
                out[(long long)s * ND + nb1 * 16 + col] = acc1[i] * scale;
            }
        }
    }
}

// ---------------------------------------------------------------------------
extern "C" void kernel_launch(void* const* d_in, const int* in_sizes, int n_in,
                              void* d_out, int out_size, void* d_ws, size_t ws_size,
                              hipStream_t stream) {
    const int*   inputs     = (const int*)d_in[0];
    const int*   node_types = (const int*)d_in[1];
    const int*   node_neigh = (const int*)d_in[2];
    const float* node_emb   = (const float*)d_in[3];
    const float* nte        = (const float*)d_in[4];
    const float* tw         = (const float*)d_in[5];
    const float* tw_s1      = (const float*)d_in[6];
    const float* tw_s2      = (const float*)d_in[7];
    (void)in_sizes; (void)n_in; (void)out_size; (void)ws_size;

    char* ws = (char*)d_ws;
    unsigned short* s1p = (unsigned short*)(ws + S1P_OFF);
    unsigned short* wp  = (unsigned short*)(ws + WP_OFF);
    unsigned* cursors   = (unsigned*)(ws + CUR_OFF);
    int* tile_type      = (int*)(ws + TT_OFF);
    int* order          = (int*)(ws + ORD_OFF);

    pack_weights<<<320, 256, 0, stream>>>(tw_s1, tw, s1p, wp);
    hist_kernel<<<1, 256, 0, stream>>>(node_types, cursors, tile_type, order);
    scatter_kernel<<<(NB + 255) / 256, 256, 0, stream>>>(node_types, cursors, order);
    gatne_tile_kernel<<<NTILES, 128, 0, stream>>>(inputs, node_neigh, node_emb, nte,
                                                  tw_s2, s1p, wp, tile_type, order,
                                                  (float*)d_out);
}
